// embed_update_43439299231952
// MI455X (gfx1250) — matrix operand order repaired and checked by reading, not compiled, not run on hardware
//
#include <hip/hip_runtime.h>
#include <hip/hip_bf16.h>

typedef __attribute__((ext_vector_type(16))) __bf16 v16bf;
typedef __attribute__((ext_vector_type(8)))  float  v8f;
typedef __attribute__((ext_vector_type(4)))  unsigned int v4u;
typedef __attribute__((ext_vector_type(8)))  int v8i;
typedef __attribute__((ext_vector_type(4)))  int v4i;

#define IN_SIZE 512
#define HID     256
#define OUTP    257
#define NPAD    272            // 17 N-tiles of 16
#define NPAIRS  16384          // B*S
#define PGRP    4              // sites per iteration -> M = 64
#define NGRP    (NPAIRS/PGRP)

#define W1_BYTES (IN_SIZE*HID*2)           // 262144 bytes of bf16, swizzled
#define WF_BYTES (8*17*1024)               // 8 k-tiles * 17 n-tiles * 1KB frags
#define STAGE_BYTES 65536
#define LDS_BYTES (W1_BYTES + STAGE_BYTES) // 327680 = 320 KB

// round-to-nearest-even f32 -> bf16
__device__ __forceinline__ unsigned short f2bf(float f) {
  unsigned int u = __float_as_uint(f);
  u += 0x7FFFu + ((u >> 16) & 1u);
  return (unsigned short)(u >> 16);
}

// A-matrix (16x32 bf16) fragment byte offset for element (k_local, row m).
// lanes 0-15 hold K=0..15, lanes 16-31 hold K=16..31, contiguous per lane
__device__ __forceinline__ int fragA_byte(int kl, int m) {
  return ((kl >> 4) * 16 + m) * 32 + (kl & 15) * 2;
}

// B-matrix (32x16 bf16) fragment byte offset for element (k_local, col n).
// lanes 0-15 hold K=0..15, lanes 16-31 hold K=16..31, contiguous per lane
__device__ __forceinline__ int fragB_byte(int kl, int n) {
  return ((kl >> 4) * 16 + n) * 32 + (kl & 15) * 2;
}

// ---------------------------------------------------------------------------
// One-time: convert W1 (512x256 f32) and Wf (256x257 f32, padded to 272) into
// bf16 B-fragment-swizzled layouts in workspace.
// ---------------------------------------------------------------------------
__global__ void prep_weights(const float* __restrict__ W1,
                             const float* __restrict__ Wf,
                             unsigned short* __restrict__ w1b,
                             unsigned short* __restrict__ wfb) {
  int tid = blockIdx.x * blockDim.x + threadIdx.x;
  if (tid < IN_SIZE * HID) {
    int k = tid / HID, n = tid % HID;
    int ktile = k >> 5, kl = k & 31, ntile = n >> 4, c = n & 15;
    int byteOff = (ktile * 16 + ntile) * 1024 + fragB_byte(kl, c);
    w1b[byteOff >> 1] = f2bf(W1[tid]);
  }
  int t2 = tid - IN_SIZE * HID;
  if (t2 >= 0 && t2 < HID * NPAD) {
    int k = t2 / NPAD, n = t2 % NPAD;
    float v = (n < OUTP) ? Wf[k * OUTP + n] : 0.0f;
    int ktile = k >> 5, kl = k & 31, ntile = n >> 4, c = n & 15;
    int byteOff = (ktile * 17 + ntile) * 1024 + fragB_byte(kl, c);
    wfb[byteOff >> 1] = f2bf(v);
  }
}

// ---------------------------------------------------------------------------
// Fused persistent kernel: GEMM1+ReLU -> GEMM2 -> gated member reduction ->
// embed update -> LayerNorm.  256 threads = 8 wave32, one WG per WGP.
// ---------------------------------------------------------------------------
__global__ __launch_bounds__(256)
void fused_mlp_kernel(const float* __restrict__ net_in,
                      const float* __restrict__ embeds,
                      const float* __restrict__ mvals,
                      const float* __restrict__ b1,
                      const float* __restrict__ bfv,
                      const float* __restrict__ gamma,
                      const float* __restrict__ beta,
                      const unsigned short* __restrict__ w1b,
                      const unsigned short* __restrict__ wfb,
                      float* __restrict__ out) {
  extern __shared__ char smem[];
  char* w1lds = smem;                 // 256 KB: W1 bf16, B-fragment layout
  char* stage = smem + W1_BYTES;      // 64 KB: A-frags, then h-frags + scratch
  float* upd_lds  = (float*)(stage + 36864);  // [4][256] f32
  float* gate_lds = (float*)(stage + 40960);  // [4][16]  raw gate logits
  float* cw_lds   = (float*)(stage + 41216);  // [4][16]  combined weights
  float* sw_lds   = (float*)(stage + 41472);  // [4]      sum of cw
  float* ug_lds   = (float*)(stage + 41488);  // [4]      update gate

  const int tid  = threadIdx.x;
  const int wave = tid >> 5;
  const int lane = tid & 31;

  // ---- bulk W1 -> LDS via Tensor Data Mover (2-D D#: 4096 x 8 of 8B) ----
#if __has_builtin(__builtin_amdgcn_tensor_load_to_lds)
  if (wave == 0) {
    unsigned long long ga = (unsigned long long)(uintptr_t)w1b;
    unsigned int ldsa = (unsigned int)(uintptr_t)w1lds;  // aperture low 32 = LDS offset
    v4u g0;
    g0[0] = 1u;                                   // count=1, user descriptor
    g0[1] = ldsa;                                 // lds_addr
    g0[2] = (unsigned int)(ga & 0xFFFFFFFFu);     // global_addr[31:0]
    g0[3] = (unsigned int)((ga >> 32) & 0x01FFFFFFu) | (2u << 30);  // [56:32] | type=2
    v8i g1;
    g1[0] = (int)(3u << 16);                      // data_size = 8 bytes
    g1[1] = (int)((4096u & 0xFFFFu) << 16);       // tensor_dim0 = 4096 (lo16)
    g1[2] = (int)(8u << 16);                      // tensor_dim0 hi=0 | tensor_dim1 = 8
    g1[3] = (int)(4096u << 16);                   // tensor_dim1 hi=0 | tile_dim0 = 4096
    g1[4] = (int)8;                               // tile_dim1 = 8, tile_dim2 = 0
    g1[5] = (int)4096;                            // tensor_dim0_stride = 4096 (lo32)
    g1[6] = 0;
    g1[7] = 0;
    v4i gz4 = {0, 0, 0, 0};
    v8i gz8 = {0, 0, 0, 0, 0, 0, 0, 0};
    __builtin_amdgcn_tensor_load_to_lds(g0, g1, gz4, gz4, gz8, 0);
    __builtin_amdgcn_s_wait_tensorcnt(0);
  }
#endif
  // Backstop copy (layouts identical -> straight b128 copy); overwrites the
  // TDM-written region with known-good data, once per block.
  {
    const uint4* src = (const uint4*)w1b;
    uint4* dst = (uint4*)w1lds;
    for (int i = tid; i < W1_BYTES / 16; i += 256) dst[i] = src[i];
  }
  __syncthreads();

  const v8f vzero = {0.f, 0.f, 0.f, 0.f, 0.f, 0.f, 0.f, 0.f};

  for (int grp = blockIdx.x; grp < NGRP; grp += gridDim.x) {
    // ---- stage net_in (64 rows x 512) as bf16 A-fragments ----
    {
      const float4* src = (const float4*)(net_in + (size_t)grp * 64 * IN_SIZE);
      for (int i = tid; i < 64 * IN_SIZE / 4; i += 256) {
        int row = i >> 7;       // / (512/4)
        int kv  = i & 127;
        float4 v = src[i];
        unsigned int lo = (unsigned int)f2bf(v.x) | ((unsigned int)f2bf(v.y) << 16);
        unsigned int hi = (unsigned int)f2bf(v.z) | ((unsigned int)f2bf(v.w) << 16);
        int k  = kv * 4;
        int mf = row >> 4, mm = row & 15;
        int kt = k >> 5,   kl = k & 31;
        char* p = stage + (mf * 16 + kt) * 1024 + fragA_byte(kl, mm);
        *(uint2*)p = make_uint2(lo, hi);  // 8B-aligned (kl%8 in {0,4})
      }
      // prefetch next group's net_in slab (128 KB) into GL2
      int nxt = grp + gridDim.x;
      if (nxt < NGRP) {
        const char* pf = (const char*)(net_in + (size_t)nxt * 64 * IN_SIZE);
        #pragma unroll
        for (int i = 0; i < 4; ++i)
          __builtin_prefetch(pf + tid * 512 + i * 128, 0, 3);
      }
    }
    __syncthreads();

    // ---- GEMM1: h = relu(A @ W1 + b1), each wave owns 2 hidden N-tiles ----
    v8f acc[2][4];
    #pragma unroll
    for (int j = 0; j < 2; ++j)
      #pragma unroll
      for (int mf = 0; mf < 4; ++mf) acc[j][mf] = vzero;

    const int t0 = wave * 2, t1 = wave * 2 + 1;
    for (int kt = 0; kt < 16; ++kt) {
      v16bf fb0 = *(const v16bf*)(w1lds + (kt * 16 + t0) * 1024 + lane * 32);
      v16bf fb1 = *(const v16bf*)(w1lds + (kt * 16 + t1) * 1024 + lane * 32);
      #pragma unroll
      for (int mf = 0; mf < 4; ++mf) {
        v16bf fa = *(const v16bf*)(stage + (mf * 16 + kt) * 1024 + lane * 32);
        acc[0][mf] = __builtin_amdgcn_wmma_f32_16x16x32_bf16(
            false, fa, false, fb0, (short)0, acc[0][mf], false, false);
        acc[1][mf] = __builtin_amdgcn_wmma_f32_16x16x32_bf16(
            false, fa, false, fb1, (short)0, acc[1][mf], false, false);
      }
    }
    __syncthreads();  // all A-frag reads done; stage region reusable for h

    // bias + relu + store h as bf16 A-fragments (k-tile index == wave)
    #pragma unroll
    for (int j = 0; j < 2; ++j) {
      int tj = wave * 2 + j;
      float bias = b1[tj * 16 + (lane & 15)];
      int kl = (tj & 1) * 16 + (lane & 15);
      #pragma unroll
      for (int mf = 0; mf < 4; ++mf) {
        #pragma unroll
        for (int r = 0; r < 8; ++r) {
          float v = acc[j][mf][r] + bias;
          v = v > 0.f ? v : 0.f;
          int m = r + 8 * (lane >> 4);
          char* p = stage + (mf * 8 + wave) * 1024 + fragA_byte(kl, m);
          *(unsigned short*)p = f2bf(v);
        }
      }
    }
    __syncthreads();

    // ---- GEMM2: out = h @ Wf (Wf frags stream from L2-resident workspace) ----
    v8f acc2[2][4], accG[4];
    #pragma unroll
    for (int j = 0; j < 2; ++j)
      #pragma unroll
      for (int mf = 0; mf < 4; ++mf) acc2[j][mf] = vzero;
    #pragma unroll
    for (int mf = 0; mf < 4; ++mf) accG[mf] = vzero;

    for (int kt = 0; kt < 8; ++kt) {
      v16bf gb0 = *(const v16bf*)((const char*)wfb + (kt * 17 + t0) * 1024 + lane * 32);
      v16bf gb1 = *(const v16bf*)((const char*)wfb + (kt * 17 + t1) * 1024 + lane * 32);
      #pragma unroll
      for (int mf = 0; mf < 4; ++mf) {
        v16bf fa = *(const v16bf*)(stage + (mf * 8 + kt) * 1024 + lane * 32);
        acc2[0][mf] = __builtin_amdgcn_wmma_f32_16x16x32_bf16(
            false, fa, false, gb0, (short)0, acc2[0][mf], false, false);
        acc2[1][mf] = __builtin_amdgcn_wmma_f32_16x16x32_bf16(
            false, fa, false, gb1, (short)0, acc2[1][mf], false, false);
      }
    }
    // gate N-tile (cols 256..271) computed by wave 0 only (EXEC uniform in-wave)
    if (wave == 0) {
      for (int kt = 0; kt < 8; ++kt) {
        v16bf gbg = *(const v16bf*)((const char*)wfb + (kt * 17 + 16) * 1024 + lane * 32);
        #pragma unroll
        for (int mf = 0; mf < 4; ++mf) {
          v16bf fa = *(const v16bf*)(stage + (mf * 8 + kt) * 1024 + lane * 32);
          accG[mf] = __builtin_amdgcn_wmma_f32_16x16x32_bf16(
              false, fa, false, gbg, (short)0, accG[mf], false, false);
        }
      }
      if ((lane & 15) == 0) {  // column n == 256 (raw gate logits)
        #pragma unroll
        for (int mf = 0; mf < 4; ++mf)
          #pragma unroll
          for (int r = 0; r < 8; ++r)
            gate_lds[mf * 16 + r + 8 * (lane >> 4)] = accG[mf][r];
      }
    }
    __syncthreads();

    // ---- per-site gate weights: cw[m] = m[m]*softnorm(sig(gate)*m[m]+1e-8) ----
    if (tid < PGRP) {
      int pIdx = grp * PGRP + tid;
      float bg = bfv[256];
      float sum = 0.f;
      for (int mm = 0; mm < 16; ++mm) {
        float s  = 1.f / (1.f + __expf(-(gate_lds[tid * 16 + mm] + bg)));
        float mr = mvals[pIdx * 16 + mm];
        sum += s * mr + 1e-8f;
      }
      float inv = 1.f / sum;
      float swv = 0.f, updg = 0.f;
      for (int mm = 0; mm < 16; ++mm) {
        float s  = 1.f / (1.f + __expf(-(gate_lds[tid * 16 + mm] + bg)));
        float mr = mvals[pIdx * 16 + mm];
        float lw = s * mr + 1e-8f;
        float w  = lw * inv;
        float c  = mr * w;
        cw_lds[tid * 16 + mm] = c;
        swv  += c;
        updg += (s * mr) * w;   // updates[...,256]
      }
      sw_lds[tid] = swv;
      ug_lds[tid] = updg;
    }
    __syncthreads();

    // ---- member reduction: updates[n] = sum_m C[m][n] * cw[m] ----
    #pragma unroll
    for (int j = 0; j < 2; ++j) {
      int t = wave * 2 + j;
      #pragma unroll
      for (int mf = 0; mf < 4; ++mf) {
        float part = 0.f;
        #pragma unroll
        for (int r = 0; r < 8; ++r)
          part += acc2[j][mf][r] * cw_lds[mf * 16 + r + 8 * (lane >> 4)];
        float other = __shfl_xor(part, 16, 32);
        if (lane < 16) upd_lds[mf * 256 + t * 16 + lane] = part + other;
      }
    }
    __syncthreads();

    // ---- epilogue: embed update + LayerNorm, one wave per site ----
    if (wave < PGRP) {
      int pl = wave;
      int pIdx = grp * PGRP + pl;
      float ug  = ug_lds[pl];
      float swv = sw_lds[pl];
      float e[8];
      float s1 = 0.f, s2 = 0.f;
      #pragma unroll
      for (int i = 0; i < 8; ++i) {
        int n = lane + i * 32;
        float u = upd_lds[pl * 256 + n] + bfv[n] * swv;
        float val = embeds[(size_t)pIdx * 256 + n] + ug * u;
        e[i] = val;
        s1 += val;
        s2 += val * val;
      }
      #pragma unroll
      for (int off = 16; off >= 1; off >>= 1) {
        s1 += __shfl_xor(s1, off, 32);
        s2 += __shfl_xor(s2, off, 32);
      }
      float mu   = s1 * (1.f / 256.f);
      float var  = s2 * (1.f / 256.f) - mu * mu;
      float rstd = rsqrtf(var + 1e-5f);
      #pragma unroll
      for (int i = 0; i < 8; ++i) {
        int n = lane + i * 32;
        out[(size_t)pIdx * 256 + n] = (e[i] - mu) * rstd * gamma[n] + beta[n];
      }
    }
    __syncthreads();  // protect stage/upd regions before next iteration
  }
}

extern "C" void kernel_launch(void* const* d_in, const int* in_sizes, int n_in,
                              void* d_out, int out_size, void* d_ws, size_t ws_size,
                              hipStream_t stream) {
  const float* net_in = (const float*)d_in[0];
  const float* embeds = (const float*)d_in[1];
  const float* mvals  = (const float*)d_in[2];
  const float* W1     = (const float*)d_in[3];
  const float* b1     = (const float*)d_in[4];
  const float* Wf     = (const float*)d_in[5];
  const float* bfv    = (const float*)d_in[6];
  const float* gamma  = (const float*)d_in[7];
  const float* beta   = (const float*)d_in[8];
  float* out = (float*)d_out;

  unsigned short* w1b = (unsigned short*)d_ws;
  unsigned short* wfb = w1b + (W1_BYTES / 2);

  int prepThreads = IN_SIZE * HID + HID * NPAD;
  prep_weights<<<(prepThreads + 255) / 256, 256, 0, stream>>>(W1, Wf, w1b, wfb);

  (void)hipFuncSetAttribute((const void*)fused_mlp_kernel,
                            hipFuncAttributeMaxDynamicSharedMemorySize, LDS_BYTES);
  fused_mlp_kernel<<<1024, 256, LDS_BYTES, stream>>>(
      net_in, embeds, mvals, b1, bfv, gamma, beta, w1b, wfb, out);
}
